// CutCrossEntropyLoss_79843442033222
// MI455X (gfx1250) — matrix operand order repaired and checked by reading, not compiled, not run on hardware
//
#include <hip/hip_runtime.h>
#include <hip/hip_bf16.h>

#define BD 2
#define SD 2048
#define DD 1024
#define VD 50257
#define NROWS (BD * (SD - 1))          // 4094 token rows
#define NROWSP 4096
#define M_TILE 64                      // tokens per workgroup (4 WMMA tiles)
#define V_BLK 2048                     // vocab columns per grid-y block
#define NVB ((VD + V_BLK - 1) / V_BLK) // 25 vocab blocks
#define K_CHUNK 32
#define NEG_INF -3.0e38f

typedef __attribute__((ext_vector_type(16))) __bf16 bf16x16;
typedef __attribute__((ext_vector_type(8))) float v8f;

union V16 {
    uint4 q[2];
    bf16x16 v;
};

__device__ __forceinline__ unsigned short f2bf(float f) {
    unsigned int u = __float_as_uint(f);
    u += 0x7FFFu + ((u >> 16) & 1u);   // round-to-nearest-even bf16
    return (unsigned short)(u >> 16);
}

// Kernel 1: per (64-token block, 2048-vocab block) partial logsumexp + target
// logit. 8 waves, each owning a 16(vocab) x 64(token) strip. W chunks are
// software-pipelined through a double-buffered LDS tile: global loads for
// chunk kc+1 are in flight while chunk kc is converted/computed (one barrier
// per chunk), feeding 4 v_wmma_f32_16x16x32_bf16 per chunk per wave.
__global__ __launch_bounds__(256) void cce_partial_kernel(
    const float* __restrict__ emb, const float* __restrict__ weight,
    const float* __restrict__ bias, const int* __restrict__ labels,
    float* __restrict__ wsMax, float* __restrict__ wsSum,
    float* __restrict__ wsTgt)
{
    __shared__ __align__(16) unsigned short embT[DD * M_TILE];      // 128 KB [k][tok]
    __shared__ __align__(16) unsigned short Wbuf[2][128 * K_CHUNK]; // 16 KB  [vloc][k]
    __shared__ float redMax[8][M_TILE];                             // 2 KB
    __shared__ float redSum[8][M_TILE];                             // 2 KB

    const int tid  = threadIdx.x;
    const int lane = tid & 31;
    const int wv   = tid >> 5;          // wave id 0..7 -> vocab stripe
    const int m0   = blockIdx.x * M_TILE;
    const int vb   = blockIdx.y;
    const int v0b  = vb * V_BLK;

    // ---- stage 64 emb rows (shifted: embeddings[:, :-1]) transposed -> bf16 ----
    {
        const int tok = tid >> 2;       // 0..63
        const int seg = tid & 3;        // 256 consecutive k each
        const int r   = m0 + tok;
        const int k0  = seg * 256;
        if (r < NROWS) {
            const int b = r / (SD - 1);
            const int s = r - b * (SD - 1);
            const float* src = emb + ((size_t)b * SD + s) * DD + k0;
            #pragma unroll 8
            for (int i = 0; i < 64; ++i) {
                float4 f = *(const float4*)(src + i * 4);
                embT[(k0 + i * 4 + 0) * M_TILE + tok] = f2bf(f.x);
                embT[(k0 + i * 4 + 1) * M_TILE + tok] = f2bf(f.y);
                embT[(k0 + i * 4 + 2) * M_TILE + tok] = f2bf(f.z);
                embT[(k0 + i * 4 + 3) * M_TILE + tok] = f2bf(f.w);
            }
        } else {
            for (int i = 0; i < 256; ++i) embT[(k0 + i) * M_TILE + tok] = 0;
        }
    }

    // per-lane token + shifted target label, one per token tile
    const int n    = lane & 15;         // WMMA N = lane%16
    const int half = lane >> 4;
    int rRowv[4], tgtv[4];
    #pragma unroll
    for (int tt = 0; tt < 4; ++tt) {
        const int r = m0 + tt * 16 + n;
        rRowv[tt] = r;
        if (r < NROWS) {
            const int b = r / (SD - 1);
            const int s = r - b * (SD - 1);
            tgtv[tt] = labels[b * SD + s + 1];
        } else {
            tgtv[tt] = -1;
        }
    }

    float rmax[4], rsum[4];
    #pragma unroll
    for (int tt = 0; tt < 4; ++tt) { rmax[tt] = NEG_INF; rsum[tt] = 0.0f; }

    // staging role: 2 threads per vocab row, 16 consecutive k each
    const int srow = tid >> 1;
    const int shf  = tid & 1;
    const int vloc = wv * 16 + n;       // A-matrix row = vocab-local index

    for (int vt = 0; vt < V_BLK / 128; ++vt) {
        const int v0 = v0b + vt * 128;
        v8f acc[4] = {};

        const int  vrow   = v0 + srow;
        const bool vvalid = (vrow < VD);
        const float* wrow = weight + (size_t)(vvalid ? vrow : 0) * DD + shf * 16;

        // prologue: issue global loads for chunk kc = 0
        float4 f[4];
        if (vvalid) {
            #pragma unroll
            for (int i = 0; i < 4; ++i) f[i] = *(const float4*)(wrow + i * 4);
        }

        for (int kc = 0; kc < DD; kc += K_CHUNK) {
            const int buf = (kc >> 5) & 1;
            // convert staged registers -> bf16 LDS (implicit wait on loads)
            {
                unsigned short* dst = &Wbuf[buf][srow * K_CHUNK + shf * 16];
                if (vvalid) {
                    #pragma unroll
                    for (int i = 0; i < 4; ++i) {
                        dst[i * 4 + 0] = f2bf(f[i].x);
                        dst[i * 4 + 1] = f2bf(f[i].y);
                        dst[i * 4 + 2] = f2bf(f[i].z);
                        dst[i * 4 + 3] = f2bf(f[i].w);
                    }
                } else {
                    #pragma unroll
                    for (int i = 0; i < 16; ++i) dst[i] = 0;
                }
            }
            __syncthreads();

            // issue global loads for the NEXT chunk; they fly during compute
            if (vvalid && (kc + K_CHUNK < DD)) {
                const float* src = wrow + kc + K_CHUNK;
                #pragma unroll
                for (int i = 0; i < 4; ++i) f[i] = *(const float4*)(src + i * 4);
            }

            // compute: A once, 4 token tiles -> 4 WMMAs
            V16 A;
            A.q[0] = *(const uint4*)&Wbuf[buf][vloc * K_CHUNK + 8 * half];
            A.q[1] = *(const uint4*)&Wbuf[buf][vloc * K_CHUNK + 16 + 8 * half];
            const int kg = kc;          // B: lane = token n, element s = k (table order)
            #pragma unroll
            for (int tt = 0; tt < 4; ++tt) {
                V16 Bm;
                #pragma unroll
                for (int s = 0; s < 16; ++s) {
                    const int kk = (s < 8) ? (8 * half + s) : (16 + 8 * half + (s - 8));
                    ((unsigned short*)&Bm)[s] = embT[(kg + kk) * M_TILE + tt * 16 + n];
                }
                acc[tt] = __builtin_amdgcn_wmma_f32_16x16x32_bf16(
                    false, A.v, false, Bm.v, (short)0, acc[tt], false, false);
            }
        }

        // ---- epilogue: online softmax over this wave's 16-vocab stripe ----
        const int vbase = v0 + wv * 16 + 8 * half; // D layout: M = j + 8*half
        float bvals[8];
        #pragma unroll
        for (int j = 0; j < 8; ++j)
            bvals[j] = (vbase + j < VD) ? bias[vbase + j] : 0.0f;

        #pragma unroll
        for (int tt = 0; tt < 4; ++tt) {
            float vals[8];
            float tmax = NEG_INF;
            #pragma unroll
            for (int j = 0; j < 8; ++j) {
                const int vg = vbase + j;
                float x = (vg < VD) ? (acc[tt][j] + bvals[j]) : NEG_INF;
                vals[j] = x;
                tmax = fmaxf(tmax, x);
                if (vg == tgtv[tt]) wsTgt[rRowv[tt]] = x; // unique hit per row
            }
            tmax = fmaxf(tmax, __shfl_xor(tmax, 16));
            const float newmax = fmaxf(rmax[tt], tmax);
            float psum = 0.0f;
            #pragma unroll
            for (int j = 0; j < 8; ++j) {
                const int vg = vbase + j;
                psum += (vg < VD) ? __expf(vals[j] - newmax) : 0.0f;
            }
            psum += __shfl_xor(psum, 16);
            rsum[tt] = rsum[tt] * __expf(rmax[tt] - newmax) + psum;
            rmax[tt] = newmax;
        }
    }

    // ---- combine 8 waves' vocab stripes -> per-(row, vblock) partial ----
    __syncthreads();
    if (lane < 16) {
        #pragma unroll
        for (int tt = 0; tt < 4; ++tt) {
            redMax[wv][tt * 16 + n] = rmax[tt];
            redSum[wv][tt * 16 + n] = rsum[tt];
        }
    }
    __syncthreads();
    if (tid < M_TILE) {
        float M = NEG_INF;
        #pragma unroll
        for (int w2 = 0; w2 < 8; ++w2) M = fmaxf(M, redMax[w2][tid]);
        float S = 0.0f;
        #pragma unroll
        for (int w2 = 0; w2 < 8; ++w2)
            S += redSum[w2][tid] * __expf(redMax[w2][tid] - M);
        const int r = m0 + tid;
        if (r < NROWS) {
            wsMax[(size_t)r * NVB + vb] = M;
            wsSum[(size_t)r * NVB + vb] = S;
        }
    }
}

// Kernel 2: merge per-row partials across vocab blocks, mean-reduce the NLL.
__global__ __launch_bounds__(256) void cce_finalize_kernel(
    const float* __restrict__ wsMax, const float* __restrict__ wsSum,
    const float* __restrict__ wsTgt, float* __restrict__ out)
{
    __shared__ float sred[256];
    const int tid = threadIdx.x;
    float accv = 0.0f;
    for (int r = tid; r < NROWS; r += 256) {
        float M = NEG_INF;
        for (int v = 0; v < NVB; ++v) M = fmaxf(M, wsMax[(size_t)r * NVB + v]);
        float S = 0.0f;
        for (int v = 0; v < NVB; ++v)
            S += wsSum[(size_t)r * NVB + v] * __expf(wsMax[(size_t)r * NVB + v] - M);
        accv += (__logf(S) + M) - wsTgt[r];
    }
    sred[tid] = accv;
    __syncthreads();
    for (int s = 128; s > 0; s >>= 1) {
        if (tid < s) sred[tid] += sred[tid + s];
        __syncthreads();
    }
    if (tid == 0) out[0] = sred[0] / (float)NROWS;
}

extern "C" void kernel_launch(void* const* d_in, const int* in_sizes, int n_in,
                              void* d_out, int out_size, void* d_ws, size_t ws_size,
                              hipStream_t stream) {
    (void)in_sizes; (void)n_in; (void)out_size; (void)ws_size;
    const float* emb    = (const float*)d_in[0];
    const float* weight = (const float*)d_in[1];
    const float* bias   = (const float*)d_in[2];
    const int*   labels = (const int*)d_in[3];

    float* wsMax = (float*)d_ws;                       // NROWSP * NVB
    float* wsSum = wsMax + (size_t)NROWSP * NVB;       // NROWSP * NVB
    float* wsTgt = wsSum + (size_t)NROWSP * NVB;       // NROWSP

    dim3 grid((NROWS + M_TILE - 1) / M_TILE, NVB);     // 64 x 25
    cce_partial_kernel<<<grid, 256, 0, stream>>>(emb, weight, bias, labels,
                                                 wsMax, wsSum, wsTgt);
    cce_finalize_kernel<<<1, 256, 0, stream>>>(wsMax, wsSum, wsTgt, (float*)d_out);
}
